// Net_20401094656098
// MI455X (gfx1250) — matrix operand order repaired and checked by reading, not compiled, not run on hardware
//
#include <hip/hip_runtime.h>

// ============================================================================
// MI455X (gfx1250) point-cloud network.
// - v_wmma_f32_16x16x32_f16 for all large contractions (f16 in, f32 acc)
// - f16 weight/activation mirrors so GEMM tiles are staged with
//   global_load_async_to_lds_b128 (ASYNCcnt) instead of load+cvt+ds_store
// ============================================================================

#define USE_ASYNC 1

typedef __attribute__((ext_vector_type(16))) _Float16 v16h;
typedef __attribute__((ext_vector_type(8)))  float    v8f;
typedef __attribute__((ext_vector_type(8)))  _Float16 h8v;

#define BB  8
#define NNP 1024
#define BN  (BB * NNP)        // 8192 points
#define KNB 16                // neighbors
#define EE  (BN * KNB)        // 131072 edges
#define BNEPS 1e-5f

__device__ __forceinline__ void async_b128_to_lds(const void* g, void* l) {
#if USE_ASYNC
  unsigned lds = (unsigned)(unsigned long long)l;   // low 32 bits = LDS offset
  asm volatile("global_load_async_to_lds_b128 %0, %1, off"
               :: "v"(lds), "v"(g) : "memory");
#else
  *(h8v*)l = *(const h8v*)g;
#endif
}
__device__ __forceinline__ void wait_asynccnt0() {
#if USE_ASYNC
  asm volatile("s_wait_asynccnt 0" ::: "memory");
#endif
}

// ---------------------------------------------------------------------------
// small utility kernels
// ---------------------------------------------------------------------------
__global__ void k_f32tof16(const float* __restrict__ s, _Float16* __restrict__ d, int n) {
  int i = blockIdx.x * blockDim.x + threadIdx.x;
  if (i < n) d[i] = (_Float16)s[i];
}

__global__ void k_concat6(const float* __restrict__ pos, const float* __restrict__ nrm,
                          float* __restrict__ x0) {
  int idx = blockIdx.x * blockDim.x + threadIdx.x;
  if (idx >= BN * 6) return;
  int p = idx / 6, c = idx % 6;
  x0[idx] = (c < 3) ? pos[p * 3 + c] : nrm[p * 3 + (c - 3)];
}

__global__ void k_concat2(const float* __restrict__ A, int Ca,
                          const float* __restrict__ Bv, int Cb,
                          float* __restrict__ O, _Float16* __restrict__ Oh) {
  int C = Ca + Cb;
  int idx = blockIdx.x * blockDim.x + threadIdx.x;
  if (idx >= BN * C) return;
  int p = idx / C, c = idx % C;
  float v = (c < Ca) ? A[(size_t)p * Ca + c] : Bv[(size_t)p * Cb + (c - Ca)];
  O[idx] = v;
  if (Oh) Oh[idx] = (_Float16)v;
}

// ---------------------------------------------------------------------------
// kNN: top-16 smallest d2 per point (within batch), ascending, stable ties.
// ---------------------------------------------------------------------------
__global__ __launch_bounds__(256) void k_knn(const float* __restrict__ pos,
                                             int* __restrict__ sid,
                                             float* __restrict__ d2o) {
  __shared__ float sx[NNP], sy[NNP], sz[NNP], sq[NNP];
  const int b  = blockIdx.x >> 2;
  const int tl = (blockIdx.x & 3) * 256 + threadIdx.x;
  for (int j = threadIdx.x; j < NNP; j += 256) {
    float x = pos[(b * NNP + j) * 3 + 0];
    float y = pos[(b * NNP + j) * 3 + 1];
    float z = pos[(b * NNP + j) * 3 + 2];
    sx[j] = x; sy[j] = y; sz[j] = z; sq[j] = x * x + y * y + z * z;
  }
  __syncthreads();
  const float px = sx[tl], py = sy[tl], pz = sz[tl], pq = sq[tl];
  float bd[KNB]; int bi[KNB];
#pragma unroll
  for (int s = 0; s < KNB; ++s) { bd[s] = 3.4e38f; bi[s] = 0; }
  for (int j = 0; j < NNP; ++j) {
    float d = pq + sq[j] - 2.f * (px * sx[j] + py * sy[j] + pz * sz[j]);
    if (d < bd[KNB - 1]) {
      int gj = b * NNP + j;
#pragma unroll
      for (int s = KNB - 1; s > 0; --s) {
        if (bd[s] > d) {
          bool up = bd[s - 1] > d;
          bd[s] = up ? bd[s - 1] : d;
          bi[s] = up ? bi[s - 1] : gj;
        }
      }
      if (bd[0] > d) { bd[0] = d; bi[0] = gj; }
    }
  }
  const int e0 = (b * NNP + tl) * KNB;
#pragma unroll
  for (int s = 0; s < KNB; ++s) { sid[e0 + s] = bi[s]; d2o[e0 + s] = bd[s]; }
}

// ---------------------------------------------------------------------------
// WMMA GEMM: Out[M x Nc] (+)= Arow[M x Kd] * W[Nc x Kd]^T (+ bias)
//   A, W are f16 (mirrors). EDGE: A row e = esign*(Xh[s]-Xh[t]).
// Block 256 thr (8 wave32). Tile 128x64, K-step 32.
// Rows mode: A and B tiles staged via global_load_async_to_lds_b128.
// Edge mode: A gathered+subtracted (v_pk_sub_f16), B async.
// ---------------------------------------------------------------------------
template <bool EDGE, bool ADDTO, bool HASBIAS>
__global__ __launch_bounds__(256) void k_gemm_wmma(
    const _Float16* __restrict__ Ah, const _Float16* __restrict__ Xh,
    const _Float16* __restrict__ Wh, const float* __restrict__ bias,
    float* __restrict__ Out, int M, int Kd, int Nc,
    const int* __restrict__ sid, const float* __restrict__ d2,
    float r2, float esign) {
  __shared__ __align__(16) _Float16 As[128][40];
  __shared__ __align__(16) _Float16 Bs[64][40];
  const int m0 = blockIdx.x * 128;
  const int n0 = blockIdx.y * 64;
  const int t = threadIdx.x;
  const int lane = t & 31, wave = t >> 5;
  const int lrA = t >> 1, lcA = (t & 1) * 16;   // 128 rows x 32 cols, 16 f16/thr
  const int lrB = t >> 2, lcB = (t & 3) * 8;    // 64 rows x 32 cols,  8 f16/thr

  const _Float16 *pS = nullptr, *pT = nullptr, *rowA = nullptr;
  if constexpr (EDGE) {
    int e = m0 + lrA;
    int gt = e >> 4;
    int gs = sid[e];
    gs = (d2[e] > r2) ? gt : gs;                // branchless radius replacement
    pS = Xh + (size_t)gs * Kd;
    pT = Xh + (size_t)gt * Kd;
  } else {
    rowA = Ah + (size_t)(m0 + lrA) * Kd;
  }
  const _Float16* rowB = Wh + (size_t)(n0 + lrB) * Kd;
  const _Float16 es = (_Float16)esign;

  v8f acc[4] = {{}, {}, {}, {}};

  for (int k0 = 0; k0 < Kd; k0 += 32) {
    if constexpr (EDGE) {
      __builtin_prefetch(pS + k0 + 32 + lcA, 0, 1);   // speculative, non-faulting
      __builtin_prefetch(pT + k0 + 32 + lcA, 0, 1);
      const h8v* ps = (const h8v*)(pS + k0 + lcA);
      const h8v* pt = (const h8v*)(pT + k0 + lcA);
      h8v e0 = (ps[0] - pt[0]) * es;
      h8v e1 = (ps[1] - pt[1]) * es;
      *(h8v*)&As[lrA][lcA]     = e0;
      *(h8v*)&As[lrA][lcA + 8] = e1;
    } else {
      async_b128_to_lds(rowA + k0 + lcA,     &As[lrA][lcA]);
      async_b128_to_lds(rowA + k0 + lcA + 8, &As[lrA][lcA + 8]);
    }
    async_b128_to_lds(rowB + k0 + lcB, &Bs[lrB][lcB]);
    wait_asynccnt0();
    __syncthreads();
    const int hf = lane >> 4, lr = lane & 15;
    v16h af;
    const _Float16* ra = &As[wave * 16 + lr][0];
#pragma unroll
    for (int i = 0; i < 8; ++i) af[i]     = ra[hf * 8 + i];
#pragma unroll
    for (int i = 0; i < 8; ++i) af[8 + i] = ra[16 + hf * 8 + i];
#pragma unroll
    for (int nj = 0; nj < 4; ++nj) {
      v16h bf;
      const _Float16* rb = &Bs[nj * 16 + lr][0];
#pragma unroll
      for (int i = 0; i < 8; ++i) bf[i]     = rb[hf * 8 + i];
#pragma unroll
      for (int i = 0; i < 8; ++i) bf[8 + i] = rb[16 + hf * 8 + i];
      acc[nj] = __builtin_amdgcn_wmma_f32_16x16x32_f16(false, af, false, bf,
                                                       (short)0, acc[nj], false, false);
    }
    __syncthreads();
  }
  const int hf = lane >> 4, ln = lane & 15;
#pragma unroll
  for (int nj = 0; nj < 4; ++nj) {
#pragma unroll
    for (int r = 0; r < 8; ++r) {
      int gm = m0 + wave * 16 + hf * 8 + r;
      int gn = n0 + nj * 16 + ln;
      float v = acc[nj][r];
      if constexpr (HASBIAS) v += bias[gn];
      size_t o = (size_t)gm * Nc + gn;
      if constexpr (ADDTO) Out[o] += v; else Out[o] = v;
    }
  }
}

// ---------------------------------------------------------------------------
// Scalar GEMM for tiny/odd shapes (lin1 K=6, fc layers M=8, fc3 N=40)
// ---------------------------------------------------------------------------
__global__ void k_gemm_scalar(const float* __restrict__ A, const float* __restrict__ W,
                              const float* __restrict__ bias, float* __restrict__ Out,
                              int M, int Kd, int Nc) {
  int idx = blockIdx.x * blockDim.x + threadIdx.x;
  if (idx >= M * Nc) return;
  int m = idx / Nc, n = idx % Nc;
  float acc = bias ? bias[n] : 0.f;
  for (int k = 0; k < Kd; ++k) acc += A[(size_t)m * Kd + k] * W[(size_t)n * Kd + k];
  Out[idx] = acc;
}

// ---------------------------------------------------------------------------
// pp1 (first=True): edge = [p[s]-p[t], nr[s]] (6) -> 64
// ---------------------------------------------------------------------------
__global__ void k_pp1_edge(const float* __restrict__ pos, const float* __restrict__ nrm,
                           const int* __restrict__ sid,
                           const float* __restrict__ W, const float* __restrict__ b,
                           float* __restrict__ Y) {
  int e = blockIdx.x * blockDim.x + threadIdx.x;
  if (e >= EE) return;
  int t = e >> 4, s = sid[e];
  float ef[6];
  ef[0] = pos[s * 3 + 0] - pos[t * 3 + 0];
  ef[1] = pos[s * 3 + 1] - pos[t * 3 + 1];
  ef[2] = pos[s * 3 + 2] - pos[t * 3 + 2];
  ef[3] = nrm[s * 3 + 0]; ef[4] = nrm[s * 3 + 1]; ef[5] = nrm[s * 3 + 2];
  for (int o = 0; o < 64; ++o) {
    float acc = b[o];
#pragma unroll
    for (int c = 0; c < 6; ++c) acc += ef[c] * W[o * 6 + c];
    Y[(size_t)e * 64 + o] = acc;
  }
}

// ---------------------------------------------------------------------------
// BatchNorm helpers: coalesced stats (64 ch x 4 rows per block) + atomics
// ---------------------------------------------------------------------------
__global__ void k_zero(float* __restrict__ p, int n) {
  int i = blockIdx.x * blockDim.x + threadIdx.x;
  if (i < n) p[i] = 0.f;
}

__global__ __launch_bounds__(256) void k_bn_stats(const float* __restrict__ X, int M, int C,
                                                  float* __restrict__ sums,
                                                  float* __restrict__ sqs) {
  const int c  = blockIdx.x * 64 + (threadIdx.x & 63);
  const int tr = threadIdx.x >> 6;
  const int stride = gridDim.y * 4;
  float s = 0.f, q = 0.f;
  for (int r = blockIdx.y * 4 + tr; r < M; r += stride) {
    float v = X[(size_t)r * C + c];
    s += v; q += v * v;
  }
  __shared__ float ss[256], qq[256];
  ss[threadIdx.x] = s; qq[threadIdx.x] = q;
  __syncthreads();
  if (tr == 0) {
    int i = threadIdx.x;
    s = ss[i] + ss[i + 64] + ss[i + 128] + ss[i + 192];
    q = qq[i] + qq[i + 64] + qq[i + 128] + qq[i + 192];
    atomicAdd(&sums[c], s);
    atomicAdd(&sqs[c], q);
  }
}

__global__ void k_bn_finalize(const float* __restrict__ sums, const float* __restrict__ sqs,
                              const float* __restrict__ g, const float* __restrict__ be,
                              float invM, int C,
                              float* __restrict__ scale, float* __restrict__ shift) {
  int c = blockIdx.x * blockDim.x + threadIdx.x;
  if (c >= C) return;
  float m = sums[c] * invM;
  float v = sqs[c] * invM - m * m;
  float sc = g[c] * rsqrtf(v + BNEPS);
  scale[c] = sc;
  shift[c] = be[c] - m * sc;
}

// normalize + relu; dual-writes f32 and (optional) f16 mirror
__global__ void k_bn_apply_relu(const float* __restrict__ X, int total, int C,
                                const float* __restrict__ sc, const float* __restrict__ sh,
                                float* __restrict__ Y, _Float16* __restrict__ Yh) {
  int idx = blockIdx.x * blockDim.x + threadIdx.x;
  if (idx >= total) return;
  int c = idx % C;
  float v = fmaxf(X[idx] * sc[c] + sh[c], 0.f);
  Y[idx] = v;
  if (Yh) Yh[idx] = (_Float16)v;
}

// normalize + relu + max over 16 neighbor slots: E x C -> BN x C (+ f16 mirror)
__global__ void k_bn_relu_maxk(const float* __restrict__ Y,
                               const float* __restrict__ sc, const float* __restrict__ sh,
                               int C, float* __restrict__ out, _Float16* __restrict__ outh) {
  int idx = blockIdx.x * blockDim.x + threadIdx.x;
  if (idx >= BN * C) return;
  int p = idx / C, c = idx % C;
  float s = sc[c], h = sh[c];
  float m = -3.4e38f;
  for (int k = 0; k < KNB; ++k) {
    float v = Y[(size_t)(p * KNB + k) * C + c] * s + h;
    m = fmaxf(m, fmaxf(v, 0.f));
  }
  out[(size_t)p * C + c] = m;
  if (outh) outh[(size_t)p * C + c] = (_Float16)m;
}

// ---------------------------------------------------------------------------
// geoconv pieces
// ---------------------------------------------------------------------------
// pack Wl(6,Cin,64) f32 -> Wpk(384, Cin) f16, row n = l*64+h
__global__ void k_pack_wl(const float* __restrict__ Wl, int Cin, _Float16* __restrict__ Wpk) {
  int idx = blockIdx.x * blockDim.x + threadIdx.x;
  if (idx >= 384 * Cin) return;
  int n = idx / Cin, c = idx % Cin;
  int l = n >> 6, h = n & 63;
  Wpk[idx] = (_Float16)Wl[((size_t)l * Cin + c) * 64 + h];
}

__global__ void k_geo_combine(const float* __restrict__ lo, const float* __restrict__ bl,
                              const float* __restrict__ pos, const int* __restrict__ sid,
                              const float* __restrict__ d2, float r2,
                              float* __restrict__ eC, float* __restrict__ pd) {
  int idx = blockIdx.x * blockDim.x + threadIdx.x;
  if (idx >= EE * 64) return;
  int e = idx >> 6, h = idx & 63;
  int t = e >> 4, s = sid[e];
  if (d2[e] > r2) s = t;
  float dx = pos[s * 3 + 0] - pos[t * 3 + 0];
  float dy = pos[s * 3 + 1] - pos[t * 3 + 1];
  float dz = pos[s * 3 + 2] - pos[t * 3 + 2];
  float dist = sqrtf(dx * dx + dy * dy + dz * dz);
  dist = fmaxf(dist, 1e-16f);
  float c0 = __cosf(dx / dist); c0 *= c0;
  float c1 = __cosf(dy / dist); c1 *= c1;
  float c2 = __cosf(dz / dist); c2 *= c2;
  int l0 = (dx > 0.f) ? 1 : 0;
  int l1 = 2 + ((dy > 0.f) ? 1 : 0);
  int l2 = 4 + ((dz > 0.f) ? 1 : 0);
  const float* L = lo + (size_t)e * 384;
  float v = (L[l0 * 64 + h] + bl[l0 * 64 + h]) * c0 +
            (L[l1 * 64 + h] + bl[l1 * 64 + h]) * c1 +
            (L[l2 * 64 + h] + bl[l2 * 64 + h]) * c2;
  eC[(size_t)e * 64 + h] = v;
  if (h == 0) pd[e] = dist;
}

__global__ void k_geo_wsum(const float* __restrict__ eC, const float* __restrict__ pd,
                           float* __restrict__ y) {
  int idx = blockIdx.x * blockDim.x + threadIdx.x;
  if (idx >= BN * 64) return;
  int p = idx >> 6, h = idx & 63;
  float mx = 0.f;
  for (int k = 0; k < KNB; ++k) mx = fmaxf(mx, pd[p * KNB + k]);
  float pr = mx * 1.1f;
  float ws = 0.f, acc = 0.f;
  for (int k = 0; k < KNB; ++k) {
    float w = pr - pd[p * KNB + k]; w = w * w;
    acc += eC[(size_t)(p * KNB + k) * 64 + h] * w;
    ws += w;
  }
  y[idx] = acc / ws;
}

__global__ void k_max_over_n(const float* __restrict__ X, int C, float* __restrict__ Y) {
  int idx = blockIdx.x * blockDim.x + threadIdx.x;
  if (idx >= BB * C) return;
  int b = idx / C, c = idx % C;
  float m = -3.4e38f;
  for (int r = 0; r < NNP; ++r) m = fmaxf(m, X[(size_t)(b * NNP + r) * C + c]);
  Y[idx] = m;
}

// ===========================================================================
// Host orchestration
// ===========================================================================
extern "C" void kernel_launch(void* const* d_in, const int* in_sizes, int n_in,
                              void* d_out, int out_size, void* d_ws, size_t ws_size,
                              hipStream_t stream) {
  (void)in_sizes; (void)n_in; (void)out_size; (void)ws_size;
  const float* pos = (const float*)d_in[0];
  const float* nrm = (const float*)d_in[1];
  enum {
    PP1_W = 2, PP1_B, PP1_G, PP1_BE,
    PP2_W, PP2_B, PP2_G, PP2_BE,
    PP3_W, PP3_B, PP3_G, PP3_BE,
    L1_W, L1_B, L1_G, L1_BE,
    C1_WL, C1_BL, C1_W1, C1_B1, C1_W2, C1_B2, C1_G1, C1_BE1, C1_G2, C1_BE2,
    L2_W, L2_B, L2_G, L2_BE,
    C2_WL, C2_BL, C2_W1, C2_B1, C2_W2, C2_B2, C2_G1, C2_BE1, C2_G2, C2_BE2,
    C3_WL, C3_BL, C3_W1, C3_B1, C3_W2, C3_B2, C3_G1, C3_BE1, C3_G2, C3_BE2,
    L3_W, L3_B, L3_G, L3_BE,
    F1_W, F1_B, F1_G, F1_BE,
    F2_W, F2_B, F2_G, F2_BE,
    F3_W, F3_B
  };
  auto P = [&](int i) { return (const float*)d_in[i]; };

  // ---- workspace layout ----
  char* base = (char*)d_ws;
  size_t off = 0;
  auto alloc = [&](size_t bytes) -> void* {
    void* p = (void*)(base + off);
    off += (bytes + 255) & ~(size_t)255;
    return p;
  };
  int*   knn_sid = (int*)alloc((size_t)EE * 4);
  float* knn_d2  = (float*)alloc((size_t)EE * 4);
  float* bigE    = (float*)alloc((size_t)EE * 384 * 4);
  float* eCb     = (float*)alloc((size_t)EE * 64 * 4);
  float* pd      = (float*)alloc((size_t)EE * 4);
  float* x0  = (float*)alloc((size_t)BN * 6 * 4);
  float* x1  = (float*)alloc((size_t)BN * 64 * 4);
  float* x2  = (float*)alloc((size_t)BN * 128 * 4);
  float* x3  = (float*)alloc((size_t)BN * 384 * 4);
  float* x4  = (float*)alloc((size_t)BN * 64 * 4);
  float* x5  = (float*)alloc((size_t)BN * 128 * 4);
  float* x6  = (float*)alloc((size_t)BN * 256 * 4);
  float* x7  = (float*)alloc((size_t)BN * 512 * 4);
  float* x8  = (float*)alloc((size_t)BN * 896 * 4);
  float* x9  = (float*)alloc((size_t)BN * 768 * 4);
  float* x10 = (float*)alloc((size_t)BN * 2048 * 4);
  float* ytmp = (float*)alloc((size_t)BN * 64 * 4);
  float* xg = (float*)alloc(8 * 2048 * 4);
  float* h1 = (float*)alloc(8 * 512 * 4);
  float* h2 = (float*)alloc(8 * 256 * 4);
  float* stats = (float*)alloc(4 * 2048 * 4);
  float* sums = stats, *sqs = stats + 2048, *scl = stats + 4096, *sft = stats + 6144;
  // f16 mirrors (activations)
  _Float16* x1h  = (_Float16*)alloc((size_t)BN * 64 * 2);
  _Float16* x2h  = (_Float16*)alloc((size_t)BN * 128 * 2);
  _Float16* x4h  = (_Float16*)alloc((size_t)BN * 64 * 2);
  _Float16* x5h  = (_Float16*)alloc((size_t)BN * 128 * 2);
  _Float16* x6h  = (_Float16*)alloc((size_t)BN * 256 * 2);
  _Float16* x8h  = (_Float16*)alloc((size_t)BN * 896 * 2);
  _Float16* x9h  = (_Float16*)alloc((size_t)BN * 768 * 2);
  _Float16* ytmph = (_Float16*)alloc((size_t)BN * 64 * 2);
  // f16 mirrors (weights)
  _Float16* hPP2 = (_Float16*)alloc(128 * 64 * 2);
  _Float16* hPP3 = (_Float16*)alloc(384 * 128 * 2);
  _Float16* hC1W1 = (_Float16*)alloc(128 * 64 * 2);
  _Float16* hC1W2 = (_Float16*)alloc(128 * 64 * 2);
  _Float16* hL2  = (_Float16*)alloc(256 * 128 * 2);
  _Float16* hC2W1 = (_Float16*)alloc(512 * 256 * 2);
  _Float16* hC2W2 = (_Float16*)alloc(512 * 64 * 2);
  _Float16* hC3W1 = (_Float16*)alloc((size_t)768 * 896 * 2);
  _Float16* hC3W2 = (_Float16*)alloc(768 * 64 * 2);
  _Float16* hL3  = (_Float16*)alloc((size_t)2048 * 768 * 2);
  _Float16* Whl  = (_Float16*)alloc((size_t)384 * 896 * 2);

  auto pack = [&](int pi, _Float16* dst, int n) {
    k_f32tof16<<<(n + 255) / 256, 256, 0, stream>>>(P(pi), dst, n);
  };
  pack(PP2_W, hPP2, 128 * 64);
  pack(PP3_W, hPP3, 384 * 128);
  pack(C1_W1, hC1W1, 128 * 64);
  pack(C1_W2, hC1W2, 128 * 64);
  pack(L2_W, hL2, 256 * 128);
  pack(C2_W1, hC2W1, 512 * 256);
  pack(C2_W2, hC2W2, 512 * 64);
  pack(C3_W1, hC3W1, 768 * 896);
  pack(C3_W2, hC3W2, 768 * 64);
  pack(L3_W, hL3, 2048 * 768);

  auto bn = [&](const float* X, int M, int C, const float* g, const float* be) {
    k_zero<<<(2 * 2048 + 255) / 256, 256, 0, stream>>>(stats, 2 * 2048);
    int rb = (M + 1023) / 1024; if (rb > 128) rb = 128; if (rb < 1) rb = 1;
    dim3 gs2(C / 64, rb);
    k_bn_stats<<<gs2, 256, 0, stream>>>(X, M, C, sums, sqs);
    k_bn_finalize<<<(C + 63) / 64, 64, 0, stream>>>(sums, sqs, g, be, 1.f / (float)M, C, scl, sft);
  };
  auto gemmRows = [&](const _Float16* Ah, const _Float16* Wh, const float* bias, float* O,
                      int M, int Kd, int Nc, bool addTo) {
    dim3 g(M / 128, Nc / 64);
    if (addTo)
      k_gemm_wmma<false, true, true><<<g, 256, 0, stream>>>(
          Ah, nullptr, Wh, bias, O, M, Kd, Nc, nullptr, nullptr, 0.f, 1.f);
    else
      k_gemm_wmma<false, false, true><<<g, 256, 0, stream>>>(
          Ah, nullptr, Wh, bias, O, M, Kd, Nc, nullptr, nullptr, 0.f, 1.f);
  };
  auto gemmEdge = [&](const _Float16* Xh, const _Float16* Wh, const float* bias, float* O,
                      int Kd, int Nc, float r2, float sign) {
    dim3 g(EE / 128, Nc / 64);
    if (bias)
      k_gemm_wmma<true, false, true><<<g, 256, 0, stream>>>(
          nullptr, Xh, Wh, bias, O, EE, Kd, Nc, knn_sid, knn_d2, r2, sign);
    else
      k_gemm_wmma<true, false, false><<<g, 256, 0, stream>>>(
          nullptr, Xh, Wh, nullptr, O, EE, Kd, Nc, knn_sid, knn_d2, r2, sign);
  };
  const float RINF = 3.4e38f;

  // ---- stage 0 ----
  k_concat6<<<(BN * 6 + 255) / 256, 256, 0, stream>>>(pos, nrm, x0);
  k_knn<<<BB * 4, 256, 0, stream>>>(pos, knn_sid, knn_d2);

  // ---- pp1 (6 -> 64) ----
  k_pp1_edge<<<(EE + 255) / 256, 256, 0, stream>>>(pos, nrm, knn_sid, P(PP1_W), P(PP1_B), bigE);
  bn(bigE, EE, 64, P(PP1_G), P(PP1_BE));
  k_bn_relu_maxk<<<(BN * 64 + 255) / 256, 256, 0, stream>>>(bigE, scl, sft, 64, x1, x1h);

  // ---- pp2 (64 -> 128): edge = x[t]-x[s] ----
  gemmEdge(x1h, hPP2, P(PP2_B), bigE, 64, 128, RINF, -1.f);
  bn(bigE, EE, 128, P(PP2_G), P(PP2_BE));
  k_bn_relu_maxk<<<(BN * 128 + 255) / 256, 256, 0, stream>>>(bigE, scl, sft, 128, x2, x2h);

  // ---- pp3 (128 -> 384) ----
  gemmEdge(x2h, hPP3, P(PP3_B), bigE, 128, 384, RINF, -1.f);
  bn(bigE, EE, 384, P(PP3_G), P(PP3_BE));
  k_bn_relu_maxk<<<(BN * 384 + 255) / 256, 256, 0, stream>>>(bigE, scl, sft, 384, x3, nullptr);

  // ---- lin1 (6 -> 64) ----
  k_gemm_scalar<<<(BN * 64 + 255) / 256, 256, 0, stream>>>(x0, P(L1_W), P(L1_B), x4, BN, 6, 64);
  bn(x4, BN, 64, P(L1_G), P(L1_BE));
  k_bn_apply_relu<<<(BN * 64 + 255) / 256, 256, 0, stream>>>(x4, BN * 64, 64, scl, sft, x4, x4h);

  // ---- geoconv ----
  auto geoconv = [&](const _Float16* xinH, int cin, int cout, float r2,
                     int WL, int BL, const _Float16* hW1, int B1i, const _Float16* hW2, int B2i,
                     int G1, int BE1, int G2, int BE2, float* xout, _Float16* xoutH) {
    k_pack_wl<<<(384 * cin + 255) / 256, 256, 0, stream>>>(P(WL), cin, Whl);
    gemmEdge(xinH, Whl, nullptr, bigE, cin, 384, r2, 1.f);        // lo = edge @ Wl
    k_geo_combine<<<(EE * 64 + 255) / 256, 256, 0, stream>>>(bigE, P(BL), pos, knn_sid,
                                                             knn_d2, r2, eCb, pd);
    k_geo_wsum<<<(BN * 64 + 255) / 256, 256, 0, stream>>>(eCb, pd, ytmp);
    bn(ytmp, BN, 64, P(G1), P(BE1));
    k_bn_apply_relu<<<(BN * 64 + 255) / 256, 256, 0, stream>>>(ytmp, BN * 64, 64, scl, sft,
                                                               ytmp, ytmph);
    gemmRows(xinH, hW1, P(B1i), xout, BN, cin, cout, false);      // xc
    gemmRows(ytmph, hW2, P(B2i), xout, BN, 64, cout, true);       // += y @ W2
    bn(xout, BN, cout, P(G2), P(BE2));
    k_bn_apply_relu<<<(BN * cout + 255) / 256, 256, 0, stream>>>(xout, BN * cout, cout,
                                                                 scl, sft, xout, xoutH);
  };

  geoconv(x4h, 64, 128, 0.15f * 0.15f,
          C1_WL, C1_BL, hC1W1, C1_B1, hC1W2, C1_B2, C1_G1, C1_BE1, C1_G2, C1_BE2, x5, x5h);

  gemmRows(x5h, hL2, P(L2_B), x6, BN, 128, 256, false);
  bn(x6, BN, 256, P(L2_G), P(L2_BE));
  k_bn_apply_relu<<<(BN * 256 + 255) / 256, 256, 0, stream>>>(x6, BN * 256, 256, scl, sft, x6, x6h);

  geoconv(x6h, 256, 512, 0.3f * 0.3f,
          C2_WL, C2_BL, hC2W1, C2_B1, hC2W2, C2_B2, C2_G1, C2_BE1, C2_G2, C2_BE2, x7, nullptr);

  k_concat2<<<(BN * 896 + 255) / 256, 256, 0, stream>>>(x3, 384, x7, 512, x8, x8h);

  geoconv(x8h, 896, 768, 0.6f * 0.6f,
          C3_WL, C3_BL, hC3W1, C3_B1, hC3W2, C3_B2, C3_G1, C3_BE1, C3_G2, C3_BE2, x9, x9h);

  gemmRows(x9h, hL3, P(L3_B), x10, BN, 768, 2048, false);
  bn(x10, BN, 2048, P(L3_G), P(L3_BE));
  k_bn_apply_relu<<<(BN * 2048 + 255) / 256, 256, 0, stream>>>(x10, BN * 2048, 2048,
                                                               scl, sft, x10, nullptr);

  k_max_over_n<<<(BB * 2048 + 255) / 256, 256, 0, stream>>>(x10, 2048, xg);

  k_gemm_scalar<<<(8 * 512 + 255) / 256, 256, 0, stream>>>(xg, P(F1_W), P(F1_B), h1, 8, 2048, 512);
  bn(h1, 8, 512, P(F1_G), P(F1_BE));
  k_bn_apply_relu<<<(8 * 512 + 255) / 256, 256, 0, stream>>>(h1, 8 * 512, 512, scl, sft, h1, nullptr);
  k_gemm_scalar<<<(8 * 256 + 255) / 256, 256, 0, stream>>>(h1, P(F2_W), P(F2_B), h2, 8, 512, 256);
  bn(h2, 8, 256, P(F2_G), P(F2_BE));
  k_bn_apply_relu<<<(8 * 256 + 255) / 256, 256, 0, stream>>>(h2, 8 * 256, 256, scl, sft, h2, nullptr);
  k_gemm_scalar<<<(8 * 40 + 255) / 256, 256, 0, stream>>>(h2, P(F3_W), P(F3_B),
                                                          (float*)d_out, 8, 256, 40);
}
